// VQVAE_14413910245527
// MI455X (gfx1250) — matrix operand order repaired and checked by reading, not compiled, not run on hardware
//
#include <hip/hip_runtime.h>
#include <cstdint>
#include <cstddef>

// ---------------- types / constants ----------------
typedef _Float16 v16h __attribute__((ext_vector_type(16)));
typedef _Float16 v8h  __attribute__((ext_vector_type(8)));
typedef float    v8f  __attribute__((ext_vector_type(8)));

#define F_RELU_IN  1
#define F_RELU_OUT 2
#define BATCH 32

// ---------------- weight prep: f32 [Co,Ci,k] (or convT [Ci,Co,k] flip) -> f16 [Co, K=Ci*k] ----------------
__global__ void prep_weight_kernel(_Float16* __restrict__ dst, const float* __restrict__ src,
                                   int Co, int Ci, int k, int tf)
{
    int i = blockIdx.x * blockDim.x + threadIdx.x;
    int n = Co * Ci * k;
    if (i >= n) return;
    int j  = i % k;
    int r  = i / k;
    int ci = r % Ci;
    int co = r / Ci;
    float w;
    if (tf) w = src[((size_t)ci * Co + co) * k + (k - 1 - j)];   // transpose + flip (ConvTranspose)
    else    w = src[i];
    dst[i] = (_Float16)w;
}

__global__ void zero1_kernel(float* p) { *p = 0.0f; }

// ---------------- generic implicit-GEMM conv1d / convT1d via WMMA ----------------
// One wave per (time tile = blockIdx.x, batch = blockIdx.y); loops all M tiles (MT = C_out/16).
// KSZ = kernel width (compile-time), DIL==2 emulates ConvTranspose via zero-inserted input, stride 1.
// NOTE: KSZ==1 path assumes stride==1, pad==0 (true for all 1x1 convs here) -> no bounds check.
template<int MT, int KSZ, int DIL>
__global__ void __launch_bounds__(32)
conv_wmma_kernel(const float* __restrict__ X, int inCt, int inCb, int Tin,
                 const _Float16* __restrict__ W, const float* __restrict__ bias,
                 int K, int stride, int pad,
                 float* __restrict__ Y, int outCt, int outCb, int Tout,
                 const float* __restrict__ RES, int resCt, int resCb,
                 int flags)
{
    int tb   = blockIdx.x;        // time tile
    int b    = blockIdx.y;        // batch
    int lane = threadIdx.x;
    int n    = lane & 15;         // time-column within tile (N), also A row (M) index
    int hi   = lane >> 4;         // K half selector (WMMA 16-bit layout)
    int t    = (tb << 4) + n;
    int tS   = t * stride - pad;
    bool reluIn = (flags & F_RELU_IN) != 0;

    const float* Xb = X + (size_t)b * inCt * Tin + (size_t)inCb * Tin;

    v8f acc[MT];
#pragma unroll
    for (int m = 0; m < MT; ++m)
#pragma unroll
        for (int r = 0; r < 8; ++r) acc[m][r] = 0.0f;

    // running (ci, j) counters for this lane's first K index (hi*16), stepping +32 per K-step
    int K0  = hi * 16;
    int ci  = K0 / KSZ;
    int j   = K0 - ci * KSZ;
    const int dci = 32 / KSZ;
    const int dj  = 32 - dci * KSZ;

    int Ksteps = K >> 5;
    for (int ks = 0; ks < Ksteps; ++ks) {
        // ---- B fragment: 16 consecutive K values (this lane's half) for column t ----
        v16h bf;
        int cie = ci, je = j;
#pragma unroll
        for (int e = 0; e < 16; ++e) {
            int u = tS + je;
            float val;
            if (KSZ == 1 && DIL == 1) {
                // 1x1 conv, stride 1, pad 0: always in-bounds
                val = Xb[(size_t)cie * Tin + u];
            } else if (DIL == 1) {
                val = 0.0f;
                if ((unsigned)u < (unsigned)Tin) val = Xb[(size_t)cie * Tin + u];
            } else {
                val = 0.0f;
                if (u >= 0 && !(u & 1)) {
                    int uu = u >> 1;
                    if (uu < Tin) val = Xb[(size_t)cie * Tin + uu];
                }
            }
            if (reluIn) val = fmaxf(val, 0.0f);
            bf[e] = (_Float16)val;
            ++je; if (je == KSZ) { je = 0; ++cie; }
        }
        ci += dci; j += dj; if (j >= KSZ) { j -= KSZ; ++ci; }

        // ---- A fragments straight from f16 weights (two b128 loads each) + WMMA ----
        int kk = ks << 5;
        const _Float16* wl = W + kk + (hi << 4);
#pragma unroll
        for (int m = 0; m < MT; ++m) {
            int co = (m << 4) + n;
            const _Float16* wp = wl + (size_t)co * K;
            union { v16h v; v8h h[2]; } au;
            au.h[0] = *(const v8h*)(wp);        // K kk+{0..7}   (or kk+{16..23} for hi=1)
            au.h[1] = *(const v8h*)(wp + 8);    // K kk+{8..15}  (or kk+{24..31} for hi=1)
            acc[m] = __builtin_amdgcn_wmma_f32_16x16x32_f16(
                false, au.v, false, bf, (short)0, acc[m], false, false);
        }
    }

    // ---- epilogue: bias + optional residual + optional ReLU, strided NCH store ----
    bool reluOut = (flags & F_RELU_OUT) != 0;
    float* Yb = Y + (size_t)b * outCt * Tout + (size_t)outCb * Tout;
    const float* Rb = RES ? (RES + (size_t)b * resCt * Tout + (size_t)resCb * Tout) : nullptr;
#pragma unroll
    for (int m = 0; m < MT; ++m) {
#pragma unroll
        for (int r = 0; r < 8; ++r) {
            int co = (m << 4) + (hi << 3) + r;   // C/D layout: VGPR r -> M = r (+8 for hi lanes)
            float val = acc[m][r] + bias[co];
            if (Rb) val += Rb[(size_t)co * Tout + t];
            if (reluOut) val = fmaxf(val, 0.0f);
            Yb[(size_t)co * Tout + t] = val;
        }
    }
}

// ---------------- vector quantization: per (b,t) column, argmin over 512 codes ----------------
// grid = (T/256, BATCH), block = 256  (T is a multiple of 256)
__global__ void quantize_kernel(const float* __restrict__ X, int T,
                                const float* __restrict__ E,   // [64, 512]
                                float* __restrict__ Q, int qCt, int qCb,
                                float* __restrict__ diff, float invN)
{
    int t = blockIdx.x * blockDim.x + threadIdx.x;
    int b = blockIdx.y;
    const float* xc = X + (size_t)b * 64 * T + t;
    float v[64];
#pragma unroll
    for (int d = 0; d < 64; ++d) v[d] = xc[(size_t)d * T];
    float best = 3.4e38f; int bidx = 0;
    for (int k = 0; k < 512; ++k) {
        float dist = 0.0f;
#pragma unroll
        for (int d = 0; d < 64; ++d) { float df = v[d] - E[d * 512 + k]; dist = fmaf(df, df, dist); }
        if (dist < best) { best = dist; bidx = k; }
    }
    float* qc = Q + (size_t)b * qCt * T + (size_t)qCb * T + t;
#pragma unroll
    for (int d = 0; d < 64; ++d) qc[(size_t)d * T] = E[d * 512 + bidx];
    atomicAdd(diff, best * invN);
}

// ---------------- parameter index map (setup_inputs insertion order) ----------------
enum {
    IN_X = 0,
    P_ENC_B_W1, P_ENC_B_B1, P_ENC_B_W2, P_ENC_B_B2, P_ENC_B_W3, P_ENC_B_B3,
    P_ENC_B_R0_W1, P_ENC_B_R0_B1, P_ENC_B_R0_W2, P_ENC_B_R0_B2,
    P_ENC_B_R1_W1, P_ENC_B_R1_B1, P_ENC_B_R1_W2, P_ENC_B_R1_B2,
    P_ENC_T_W1, P_ENC_T_B1, P_ENC_T_W2, P_ENC_T_B2,
    P_ENC_T_R0_W1, P_ENC_T_R0_B1, P_ENC_T_R0_W2, P_ENC_T_R0_B2,
    P_ENC_T_R1_W1, P_ENC_T_R1_B1, P_ENC_T_R1_W2, P_ENC_T_R1_B2,
    P_QCONV_T_W, P_QCONV_T_B, P_QCONV_B_W, P_QCONV_B_B,
    P_EMBED_T, P_EMBED_B,
    P_DEC_T_WIN, P_DEC_T_BIN,
    P_DEC_T_R0_W1, P_DEC_T_R0_B1, P_DEC_T_R0_W2, P_DEC_T_R0_B2,
    P_DEC_T_R1_W1, P_DEC_T_R1_B1, P_DEC_T_R1_W2, P_DEC_T_R1_B2,
    P_DEC_T_WT, P_DEC_T_BT,
    P_UP_T_W, P_UP_T_B,
    P_DEC_WIN, P_DEC_BIN,
    P_DEC_R0_W1, P_DEC_R0_B1, P_DEC_R0_W2, P_DEC_R0_B2,
    P_DEC_R1_W1, P_DEC_R1_B1, P_DEC_R1_W2, P_DEC_R1_B2,
    P_DEC_WT1, P_DEC_BT1, P_DEC_WT2, P_DEC_BT2
};

static inline void conv_launch(hipStream_t s,
    const float* X, int inCt, int inCb, int Tin,
    const _Float16* W, const float* bias, int Co, int K, int ksz, int stride, int pad, int dil,
    float* Y, int outCt, int outCb, int Tout,
    const float* RES, int resCt, int resCb, int flags)
{
    dim3 grid(Tout / 16, BATCH), blk(32);
#define CONV_ARGS X, inCt, inCb, Tin, W, bias, K, stride, pad, Y, outCt, outCb, Tout, RES, resCt, resCb, flags
#define DISPATCH_MT(KSZV, DILV)                                                                   \
    switch (Co / 16) {                                                                            \
    case 2: conv_wmma_kernel<2, KSZV, DILV><<<grid, blk, 0, s>>>(CONV_ARGS); break;               \
    case 4: conv_wmma_kernel<4, KSZV, DILV><<<grid, blk, 0, s>>>(CONV_ARGS); break;               \
    case 5: conv_wmma_kernel<5, KSZV, DILV><<<grid, blk, 0, s>>>(CONV_ARGS); break;               \
    case 8: conv_wmma_kernel<8, KSZV, DILV><<<grid, blk, 0, s>>>(CONV_ARGS); break;               \
    default: break; }
    if (dil == 1) {
        if (ksz == 1)      { DISPATCH_MT(1, 1) }
        else if (ksz == 3) { DISPATCH_MT(3, 1) }
        else               { DISPATCH_MT(4, 1) }
    } else                 { DISPATCH_MT(4, 2) }
#undef DISPATCH_MT
#undef CONV_ARGS
}

extern "C" void kernel_launch(void* const* d_in, const int* in_sizes, int n_in,
                              void* d_out, int out_size, void* d_ws, size_t ws_size,
                              hipStream_t stream)
{
    (void)in_sizes; (void)n_in; (void)out_size; (void)ws_size;
    const float* x = (const float*)d_in[IN_X];
    auto BP = [&](int idx) { return (const float*)d_in[idx]; };

    // ---- workspace layout (bytes): [0,2MB) f16 weights; then f32 buffers ----
    char* wsb = (char*)d_ws;
    _Float16* wcur = (_Float16*)wsb;
    float* bufA = (float*)(wsb + (size_t)( 2u << 20));   // 16 MB: up to 32x128x1024 / 32x64x2048
    float* bufB = (float*)(wsb + (size_t)(18u << 20));   // 16 MB
    float* bufC = (float*)(wsb + (size_t)(34u << 20));   // 24 MB: concat 32x192x1024
    float* bufD = (float*)(wsb + (size_t)(58u << 20));   //  4 MB: qt 32x64x512

    // ---- d_out partition: dec [32,80,4096] | diff [1] | quant [32,128,1024] ----
    float* out_dec   = (float*)d_out;
    float* out_diff  = out_dec + (size_t)32 * 80 * 4096;
    float* out_quant = out_diff + 1;

    // ---- stage 0: convert all conv weights to f16 [Co,K] (convT: transpose+flip) ----
    auto prep = [&](int idx, int Co, int Ci, int k, bool tf) -> const _Float16* {
        _Float16* dst = wcur; int nn = Co * Ci * k;
        prep_weight_kernel<<<(nn + 255) / 256, 256, 0, stream>>>(dst, (const float*)d_in[idx], Co, Ci, k, tf ? 1 : 0);
        wcur += nn; return dst;
    };
    const _Float16 *Wb1   = prep(P_ENC_B_W1,   64,  80, 4, false);
    const _Float16 *Wb2   = prep(P_ENC_B_W2,  128,  64, 4, false);
    const _Float16 *Wb3   = prep(P_ENC_B_W3,  128, 128, 3, false);
    const _Float16 *Wbr0a = prep(P_ENC_B_R0_W1, 32, 128, 3, false);
    const _Float16 *Wbr0b = prep(P_ENC_B_R0_W2, 128, 32, 1, false);
    const _Float16 *Wbr1a = prep(P_ENC_B_R1_W1, 32, 128, 3, false);
    const _Float16 *Wbr1b = prep(P_ENC_B_R1_W2, 128, 32, 1, false);
    const _Float16 *Wt1   = prep(P_ENC_T_W1,   64, 128, 4, false);
    const _Float16 *Wt2   = prep(P_ENC_T_W2,  128,  64, 3, false);
    const _Float16 *Wtr0a = prep(P_ENC_T_R0_W1, 32, 128, 3, false);
    const _Float16 *Wtr0b = prep(P_ENC_T_R0_W2, 128, 32, 1, false);
    const _Float16 *Wtr1a = prep(P_ENC_T_R1_W1, 32, 128, 3, false);
    const _Float16 *Wtr1b = prep(P_ENC_T_R1_W2, 128, 32, 1, false);
    const _Float16 *Wqt   = prep(P_QCONV_T_W,  64, 128, 1, false);
    const _Float16 *Wqb   = prep(P_QCONV_B_W,  64, 192, 1, false);
    const _Float16 *Wdtin = prep(P_DEC_T_WIN, 128,  64, 3, false);
    const _Float16 *Wdtr0a= prep(P_DEC_T_R0_W1, 32, 128, 3, false);
    const _Float16 *Wdtr0b= prep(P_DEC_T_R0_W2, 128, 32, 1, false);
    const _Float16 *Wdtr1a= prep(P_DEC_T_R1_W1, 32, 128, 3, false);
    const _Float16 *Wdtr1b= prep(P_DEC_T_R1_W2, 128, 32, 1, false);
    const _Float16 *Wdtwt = prep(P_DEC_T_WT,   64, 128, 4, true);   // ConvT 128->64
    const _Float16 *Wupt  = prep(P_UP_T_W,     64,  64, 4, true);   // ConvT 64->64
    const _Float16 *Wdin  = prep(P_DEC_WIN,   128, 128, 3, false);
    const _Float16 *Wdr0a = prep(P_DEC_R0_W1,  32, 128, 3, false);
    const _Float16 *Wdr0b = prep(P_DEC_R0_W2, 128,  32, 1, false);
    const _Float16 *Wdr1a = prep(P_DEC_R1_W1,  32, 128, 3, false);
    const _Float16 *Wdr1b = prep(P_DEC_R1_W2, 128,  32, 1, false);
    const _Float16 *Wdw1  = prep(P_DEC_WT1,    64, 128, 4, true);   // ConvT 128->64
    const _Float16 *Wdw2  = prep(P_DEC_WT2,    80,  64, 4, true);   // ConvT 64->80

    // ================= enc_b (stride 4): x[32,80,4096] -> cat_b ch64..191 [32,192,1024] =================
    conv_launch(stream, x,    80, 0, 4096, Wb1, BP(P_ENC_B_B1),  64, 320, 4, 2, 1, 1, bufA,  64, 0, 2048, nullptr, 0, 0, F_RELU_OUT);
    conv_launch(stream, bufA, 64, 0, 2048, Wb2, BP(P_ENC_B_B2), 128, 256, 4, 2, 1, 1, bufB, 128, 0, 1024, nullptr, 0, 0, F_RELU_OUT);
    conv_launch(stream, bufB,128, 0, 1024, Wb3, BP(P_ENC_B_B3), 128, 384, 3, 1, 1, 1, bufA, 128, 0, 1024, nullptr, 0, 0, 0);
    // res0
    conv_launch(stream, bufA,128, 0, 1024, Wbr0a, BP(P_ENC_B_R0_B1),  32, 384, 3, 1, 1, 1, bufB,  32, 0, 1024, nullptr, 0, 0, F_RELU_IN);
    conv_launch(stream, bufB, 32, 0, 1024, Wbr0b, BP(P_ENC_B_R0_B2), 128,  32, 1, 1, 0, 1, bufA, 128, 0, 1024, bufA, 128, 0, F_RELU_IN);
    // res1 (+final encoder ReLU), writing into concat buffer channels 64..191
    conv_launch(stream, bufA,128, 0, 1024, Wbr1a, BP(P_ENC_B_R1_B1),  32, 384, 3, 1, 1, 1, bufB,  32, 0, 1024, nullptr, 0, 0, F_RELU_IN);
    conv_launch(stream, bufB, 32, 0, 1024, Wbr1b, BP(P_ENC_B_R1_B2), 128,  32, 1, 1, 0, 1, bufC, 192, 64, 1024, bufA, 128, 0, F_RELU_IN | F_RELU_OUT);

    // ================= enc_t (stride 2): cat_b ch64..191 -> [32,128,512] in bufB =================
    conv_launch(stream, bufC,192,64, 1024, Wt1, BP(P_ENC_T_B1),  64, 512, 4, 2, 1, 1, bufA,  64, 0, 512, nullptr, 0, 0, F_RELU_OUT);
    conv_launch(stream, bufA, 64, 0,  512, Wt2, BP(P_ENC_T_B2), 128, 192, 3, 1, 1, 1, bufB, 128, 0, 512, nullptr, 0, 0, 0);
    conv_launch(stream, bufB,128, 0,  512, Wtr0a, BP(P_ENC_T_R0_B1),  32, 384, 3, 1, 1, 1, bufA,  32, 0, 512, nullptr, 0, 0, F_RELU_IN);
    conv_launch(stream, bufA, 32, 0,  512, Wtr0b, BP(P_ENC_T_R0_B2), 128,  32, 1, 1, 0, 1, bufB, 128, 0, 512, bufB, 128, 0, F_RELU_IN);
    conv_launch(stream, bufB,128, 0,  512, Wtr1a, BP(P_ENC_T_R1_B1),  32, 384, 3, 1, 1, 1, bufA,  32, 0, 512, nullptr, 0, 0, F_RELU_IN);
    conv_launch(stream, bufA, 32, 0,  512, Wtr1b, BP(P_ENC_T_R1_B2), 128,  32, 1, 1, 0, 1, bufB, 128, 0, 512, bufB, 128, 0, F_RELU_IN | F_RELU_OUT);

    // ================= top quantizer =================
    conv_launch(stream, bufB,128, 0, 512, Wqt, BP(P_QCONV_T_B), 64, 128, 1, 1, 0, 1, bufA, 64, 0, 512, nullptr, 0, 0, 0);
    zero1_kernel<<<1, 1, 0, stream>>>(out_diff);
    quantize_kernel<<<dim3(512 / 256, BATCH), 256, 0, stream>>>(
        bufA, 512, BP(P_EMBED_T), bufD, 64, 0, out_diff, 1.0f / (32.0f * 512.0f * 64.0f));

    // ================= dec_t (stride 2): qt [32,64,512] -> cat_b ch0..63 [32,64,1024] =================
    conv_launch(stream, bufD, 64, 0, 512, Wdtin, BP(P_DEC_T_BIN), 128, 192, 3, 1, 1, 1, bufA, 128, 0, 512, nullptr, 0, 0, 0);
    conv_launch(stream, bufA,128, 0, 512, Wdtr0a, BP(P_DEC_T_R0_B1),  32, 384, 3, 1, 1, 1, bufB,  32, 0, 512, nullptr, 0, 0, F_RELU_IN);
    conv_launch(stream, bufB, 32, 0, 512, Wdtr0b, BP(P_DEC_T_R0_B2), 128,  32, 1, 1, 0, 1, bufA, 128, 0, 512, bufA, 128, 0, F_RELU_IN);
    conv_launch(stream, bufA,128, 0, 512, Wdtr1a, BP(P_DEC_T_R1_B1),  32, 384, 3, 1, 1, 1, bufB,  32, 0, 512, nullptr, 0, 0, F_RELU_IN);
    conv_launch(stream, bufB, 32, 0, 512, Wdtr1b, BP(P_DEC_T_R1_B2), 128,  32, 1, 1, 0, 1, bufA, 128, 0, 512, bufA, 128, 0, F_RELU_IN | F_RELU_OUT);
    // ConvT 128->64 k4 s2 p1  => dil=2, stride=1, pad=k-1-p=2
    conv_launch(stream, bufA,128, 0, 512, Wdtwt, BP(P_DEC_T_BT), 64, 512, 4, 1, 2, 2, bufC, 192, 0, 1024, nullptr, 0, 0, 0);

    // ================= bottom quantizer (writes directly into quant output ch64..127) =================
    conv_launch(stream, bufC,192, 0, 1024, Wqb, BP(P_QCONV_B_B), 64, 192, 1, 1, 0, 1, bufA, 64, 0, 1024, nullptr, 0, 0, 0);
    quantize_kernel<<<dim3(1024 / 256, BATCH), 256, 0, stream>>>(
        bufA, 1024, BP(P_EMBED_B), out_quant, 128, 64, out_diff, 1.0f / (32.0f * 1024.0f * 64.0f));

    // ================= up_t: ConvT(qt) 64->64 -> quant output ch0..63 =================
    conv_launch(stream, bufD, 64, 0, 512, Wupt, BP(P_UP_T_B), 64, 256, 4, 1, 2, 2, out_quant, 128, 0, 1024, nullptr, 0, 0, 0);

    // ================= dec (stride 4): quant [32,128,1024] -> dec output [32,80,4096] =================
    conv_launch(stream, out_quant,128, 0, 1024, Wdin, BP(P_DEC_BIN), 128, 384, 3, 1, 1, 1, bufA, 128, 0, 1024, nullptr, 0, 0, 0);
    conv_launch(stream, bufA,128, 0, 1024, Wdr0a, BP(P_DEC_R0_B1),  32, 384, 3, 1, 1, 1, bufB,  32, 0, 1024, nullptr, 0, 0, F_RELU_IN);
    conv_launch(stream, bufB, 32, 0, 1024, Wdr0b, BP(P_DEC_R0_B2), 128,  32, 1, 1, 0, 1, bufA, 128, 0, 1024, bufA, 128, 0, F_RELU_IN);
    conv_launch(stream, bufA,128, 0, 1024, Wdr1a, BP(P_DEC_R1_B1),  32, 384, 3, 1, 1, 1, bufB,  32, 0, 1024, nullptr, 0, 0, F_RELU_IN);
    conv_launch(stream, bufB, 32, 0, 1024, Wdr1b, BP(P_DEC_R1_B2), 128,  32, 1, 1, 0, 1, bufA, 128, 0, 1024, bufA, 128, 0, F_RELU_IN | F_RELU_OUT);
    // ConvT 128->64 k4 s2 p1 (+ReLU)
    conv_launch(stream, bufA,128, 0, 1024, Wdw1, BP(P_DEC_BT1), 64, 512, 4, 1, 2, 2, bufB, 64, 0, 2048, nullptr, 0, 0, F_RELU_OUT);
    // ConvT 64->80 k4 s2 p1 -> final output
    conv_launch(stream, bufB, 64, 0, 2048, Wdw2, BP(P_DEC_BT2), 80, 256, 4, 1, 2, 2, out_dec, 80, 0, 4096, nullptr, 0, 0, 0);
}
